// YourNet_88562225644017
// MI455X (gfx1250) — matrix operand order repaired and checked by reading, not compiled, not run on hardware
//
#include <hip/hip_runtime.h>
#include <cstddef>

typedef __attribute__((ext_vector_type(16))) _Float16 v16h;
typedef __attribute__((ext_vector_type(8)))  float    v8f;

#define DEV static __device__ __forceinline__
#define BTOT 8192
#define TMAX 100
#define NFRAG 55

DEV v8f wmma32(v16h a, v16h b, v8f c) {
  return __builtin_amdgcn_wmma_f32_16x16x32_f16(false, a, false, b, (short)0, c, false, false);
}
DEV v8f splat8(float s) {
  v8f c;
#pragma unroll
  for (int i = 0; i < 8; ++i) c[i] = s;
  return c;
}
DEV v16h zero16() {
  v16h a;
#pragma unroll
  for (int i = 0; i < 16; ++i) a[i] = (_Float16)0.f;
  return a;
}

// ---- fast nonlinearities (single-instruction TRANS ops, no libm/branches) --
DEV float fast_rcp(float x) { return __builtin_amdgcn_rcpf(x); }
DEV float fast_tanh(float x) {
#if __has_builtin(__builtin_amdgcn_tanhf)
  return __builtin_amdgcn_tanhf(x);        // V_TANH_F32
#else
  float e = __expf(2.0f * x);              // V_EXP_F32
  return 1.0f - 2.0f * fast_rcp(e + 1.0f); // V_RCP_F32
#endif
}
DEV float fast_sigmoid(float x) {
#if __has_builtin(__builtin_amdgcn_tanhf)
  return fmaf(0.5f, __builtin_amdgcn_tanhf(0.5f * x), 0.5f);
#else
  return fast_rcp(1.0f + __expf(-x));
#endif
}

// A-fragment K index map: element v -> K offset within a 32-wide K window
DEV int kmapA(int v, int khalf) { return khalf + v + (v < 8 ? 0 : 8); }

// A-frag (16 x 32 tile, M = batch rows) from row-major f16 [16][row_stride]
DEV v16h load_a_f16(const _Float16* base, int row_stride, int kofs, int ktot) {
  const int lane = threadIdx.x & 31, m = lane & 15, khalf = (lane >> 4) * 8;
  v16h a;
#pragma unroll
  for (int v = 0; v < 16; ++v) {
    int k = kofs + kmapA(v, khalf);
    a[v] = (k < ktot) ? base[m * row_stride + k] : (_Float16)0.f;
  }
  return a;
}

// B-frag (32K x 16N): B[k][n] = W[row0+n][k] from torch-layout f32 W[out][ldw]
DEV v16h load_b_f32(const float* W, int ldw, int row0, int kofs, int ktot, int nmax) {
  const int lane = threadIdx.x & 31, n = lane & 15, kb = (lane >> 4) * 8;
  v16h b;
#pragma unroll
  for (int v = 0; v < 16; ++v) {
    int k = kofs + kb + v + (v < 8 ? 0 : 8);
    float val = (k < ktot && n < nmax) ? W[(row0 + n) * ldw + k] : 0.f;
    b[v] = (_Float16)val;
  }
  return b;
}

// ------------------------------ encoders ------------------------------------
struct EncArgs {
  const float* x[4];
  int T[4];
  const float* emb_w[4]; const float* emb_b[4];
  const float* wih[4][4]; const float* whh[4][4];
  const float* bih[4][4]; const float* bhh[4][4];
  const float* dyn_w; const float* dyn_b;
  _Float16* enc_out;   // [BTOT][64] f16
};

__launch_bounds__(32)
__global__ void encoder_kernel(EncArgs A) {
  extern __shared__ _Float16 sm[];
  _Float16* seq    = sm;               // [TMAX][16][16] f16, in-place across layers
  _Float16* bounce = sm + TMAX * 256;  // [16][16] f16 D->A bounce tile

  const int e     = blockIdx.y;
  const int T     = A.T[e];
  const int bbase = blockIdx.x * 16;
  const int lane  = threadIdx.x & 31;
  const int m     = lane & 15;
  const int khalf = (lane >> 4) * 8;
  const int n     = lane & 15;
  const int mrow  = (lane >> 4) * 8;   // C/D layout row base

  const float* X  = A.x[e];
  const float* EW = A.emb_w[e];
  const float* EB = A.emb_b[e];

  v16h ah = zero16();
  v8f  hc = splat8(0.f);

  for (int l = 0; l < 4; ++l) {
    const int in = (l == 0) ? 32 : 16;
    const float* wih = A.wih[e][l];
    const float* whh = A.whh[e][l];
    v16h wr = load_b_f32(wih, in, 0, 0, in, 16);
    v16h wz = load_b_f32(wih, in, 16, 0, in, 16);
    v16h wn = load_b_f32(wih, in, 32, 0, in, 16);
    v16h ur = load_b_f32(whh, 16, 0, 0, 16, 16);
    v16h uz = load_b_f32(whh, 16, 16, 0, 16, 16);
    v16h un = load_b_f32(whh, 16, 32, 0, 16, 16);
    const float* bih = A.bih[e][l];
    const float* bhh = A.bhh[e][l];
    const float br  = bih[n] + bhh[n];
    const float bz  = bih[16 + n] + bhh[16 + n];
    const float bxn = bih[32 + n];
    const float bhn = bhh[32 + n];

    ah = zero16();
    hc = splat8(0.f);

    for (int t = 0; t < T; ++t) {
      __syncthreads();
      v16h ax;
      if (l == 0) {
        // inline embedding: relu(W32x2 * x + b) directly in A-frag layout
        const float x0 = X[((size_t)t * BTOT + bbase + m) * 2 + 0];
        const float x1 = X[((size_t)t * BTOT + bbase + m) * 2 + 1];
#pragma unroll
        for (int v = 0; v < 16; ++v) {
          int j = kmapA(v, khalf);
          float ev = fmaf(EW[j * 2 + 0], x0, fmaf(EW[j * 2 + 1], x1, EB[j]));
          ax[v] = (_Float16)fmaxf(ev, 0.f);
        }
      } else {
        ax = load_a_f16(seq + t * 256, 16, 0, 16);
      }
      v8f cR = splat8(br);  cR = wmma32(ax, wr, cR);  cR = wmma32(ah, ur, cR);
      v8f cZ = splat8(bz);  cZ = wmma32(ax, wz, cZ);  cZ = wmma32(ah, uz, cZ);
      v8f cXn = splat8(bxn); cXn = wmma32(ax, wn, cXn);
      v8f cHn = splat8(bhn); cHn = wmma32(ah, un, cHn);
#pragma unroll
      for (int i = 0; i < 8; ++i) {
        float r  = fast_sigmoid(cR[i]);
        float z  = fast_sigmoid(cZ[i]);
        float nn = fast_tanh(cXn[i] + r * cHn[i]);
        hc[i] = (1.f - z) * nn + z * hc[i];
      }
      __syncthreads();
#pragma unroll
      for (int i = 0; i < 8; ++i)
        bounce[(mrow + i) * 16 + n] = (_Float16)hc[i];
      __syncthreads();
      ah = load_a_f16(bounce, 16, 0, 16);
#pragma unroll
      for (int j = 0; j < 8; ++j)
        seq[t * 256 + m * 16 + khalf + j] = ah[j];  // layer output (in-place)
    }
  }
  // dyn_emb (16->16) + relu -> enc chunk e
  v16h bd = load_b_f32(A.dyn_w, 16, 0, 0, 16, 16);
  v8f  cd = splat8(A.dyn_b[n]);
  cd = wmma32(ah, bd, cd);
#pragma unroll
  for (int i = 0; i < 8; ++i) {
    float v = fmaxf(cd[i], 0.f);
    A.enc_out[((size_t)(bbase + mrow + i)) * 64 + e * 16 + n] = (_Float16)v;
  }
}

// ------------------------------ decoder -------------------------------------
struct DecArgs {
  const _Float16* enc;   // [BTOT][64]
  const float* wih[4]; const float* whh[4]; const float* bih[4]; const float* bhh[4];
  const float* out_w; const float* out_b;
  float* out;            // [50][BTOT][5]
};

DEV void fill_frag(_Float16* dst, const float* W, int ldw, int row0, int kofs,
                   int ktot, int nmax) {
  const int lane = threadIdx.x & 31, n = lane & 15, kb = (lane >> 4) * 8;
#pragma unroll
  for (int v = 0; v < 16; ++v) {
    int k = kofs + kb + v + (v < 8 ? 0 : 8);
    float val = (k < ktot && n < nmax) ? W[(row0 + n) * ldw + k] : 0.f;
    dst[lane * 16 + v] = (_Float16)val;
  }
}
DEV v16h read_frag(const _Float16* sm, int f) {
  const int lane = threadIdx.x & 31;
  const _Float16* p = sm + f * 512 + lane * 16;
  v16h b;
#pragma unroll
  for (int v = 0; v < 16; ++v) b[v] = p[v];
  return b;
}

__launch_bounds__(128)
__global__ void decoder_kernel(DecArgs A) {
  extern __shared__ _Float16 sm[];
  const int wave  = threadIdx.x >> 5;
  const int lane  = threadIdx.x & 31;
  const int n     = lane & 15;
  const int mrow  = (lane >> 4) * 8;
  _Float16* bounce = sm + NFRAG * 512 + wave * 512;   // per-wave 16x32 tile
  const int bbase = (blockIdx.x * 4 + wave) * 16;

  // gather all decoder weights into B-frag-ordered f16 LDS
  for (int f = wave; f < NFRAG; f += 4) {
    if (f < 24) {        // whh[l], gate g, ntile nt
      int l = f / 6, r = f % 6, g = r / 2, nt = r & 1;
      fill_frag(sm + f * 512, A.whh[l], 32, g * 32 + nt * 16, 0, 32, 16);
    } else if (f < 42) { // wih[1..3]
      int q = f - 24, l = 1 + q / 6, r = q % 6, g = r / 2, nt = r & 1;
      fill_frag(sm + f * 512, A.wih[l], 32, g * 32 + nt * 16, 0, 32, 16);
    } else if (f < 54) { // wih[0] (K=64, two K-halves)
      int q = f - 42, g = q / 4, r = q % 4, nt = r >> 1, kh = r & 1;
      fill_frag(sm + f * 512, A.wih[0], 64, g * 32 + nt * 16, kh * 32, 64, 16);
    } else {             // out head [5,32]
      fill_frag(sm + f * 512, A.out_w, 32, 0, 0, 32, 5);
    }
  }
  __syncthreads();

  float bRZ[4][2][2], bXn[4][2], bHn[4][2];
#pragma unroll
  for (int l = 0; l < 4; ++l)
#pragma unroll
    for (int nt = 0; nt < 2; ++nt) {
      bRZ[l][0][nt] = A.bih[l][nt * 16 + n] + A.bhh[l][nt * 16 + n];
      bRZ[l][1][nt] = A.bih[l][32 + nt * 16 + n] + A.bhh[l][32 + nt * 16 + n];
      bXn[l][nt]    = A.bih[l][64 + nt * 16 + n];
      bHn[l][nt]    = A.bhh[l][64 + nt * 16 + n];
    }

  // constant input projection (layer 0): xp0 = enc @ wih0^T + biases
  v16h aE0 = load_a_f16(A.enc + (size_t)bbase * 64, 64, 0, 64);
  v16h aE1 = load_a_f16(A.enc + (size_t)bbase * 64, 64, 32, 64);
  v8f xp0[3][2];
#pragma unroll
  for (int g = 0; g < 3; ++g)
#pragma unroll
    for (int nt = 0; nt < 2; ++nt) {
      int row = g * 32 + nt * 16;
      float b0 = A.bih[0][row + n] + (g < 2 ? A.bhh[0][row + n] : 0.f);
      v8f c = splat8(b0);
      c = wmma32(aE0, read_frag(sm, 42 + g * 4 + nt * 2 + 0), c);
      c = wmma32(aE1, read_frag(sm, 42 + g * 4 + nt * 2 + 1), c);
      xp0[g][nt] = c;
    }

  v16h hA[4];
  v8f  hc[4][2];
#pragma unroll
  for (int l = 0; l < 4; ++l) {
    hA[l] = zero16();
    hc[l][0] = splat8(0.f);
    hc[l][1] = splat8(0.f);
  }

  for (int t = 0; t < 50; ++t) {
    v16h xA = zero16();
#pragma unroll
    for (int l = 0; l < 4; ++l) {
#pragma unroll
      for (int nt = 0; nt < 2; ++nt) {
        v8f cR, cZ, cXn;
        if (l == 0) {
          cR = xp0[0][nt]; cZ = xp0[1][nt]; cXn = xp0[2][nt];
        } else {
          cR  = splat8(bRZ[l][0][nt]);
          cR  = wmma32(xA, read_frag(sm, 24 + (l - 1) * 6 + 0 * 2 + nt), cR);
          cZ  = splat8(bRZ[l][1][nt]);
          cZ  = wmma32(xA, read_frag(sm, 24 + (l - 1) * 6 + 1 * 2 + nt), cZ);
          cXn = splat8(bXn[l][nt]);
          cXn = wmma32(xA, read_frag(sm, 24 + (l - 1) * 6 + 2 * 2 + nt), cXn);
        }
        v8f cHn = splat8(bHn[l][nt]);
        cR  = wmma32(hA[l], read_frag(sm, l * 6 + 0 * 2 + nt), cR);
        cZ  = wmma32(hA[l], read_frag(sm, l * 6 + 1 * 2 + nt), cZ);
        cHn = wmma32(hA[l], read_frag(sm, l * 6 + 2 * 2 + nt), cHn);
#pragma unroll
        for (int i = 0; i < 8; ++i) {
          float r  = fast_sigmoid(cR[i]);
          float z  = fast_sigmoid(cZ[i]);
          float nn = fast_tanh(cXn[i] + r * cHn[i]);
          hc[l][nt][i] = (1.f - z) * nn + z * hc[l][nt][i];
        }
      }
      __syncthreads();
#pragma unroll
      for (int nt = 0; nt < 2; ++nt)
#pragma unroll
        for (int i = 0; i < 8; ++i)
          bounce[(mrow + i) * 32 + nt * 16 + n] = (_Float16)hc[l][nt][i];
      __syncthreads();
      hA[l] = load_a_f16(bounce, 32, 0, 32);
      xA = hA[l];
    }
    // output head + transforms
    v8f co = splat8(n < 5 ? A.out_b[n] : 0.f);
    co = wmma32(hA[3], read_frag(sm, 54), co);
#pragma unroll
    for (int i = 0; i < 8; ++i) {
      if (n < 5) {
        float v = co[i];
        if (n == 2 || n == 3) v = __expf(v);
        else if (n == 4)      v = fast_tanh(v);
        A.out[((size_t)t * BTOT + bbase + mrow + i) * 5 + n] = v;
      }
    }
  }
}

// ------------------------------ host launch ---------------------------------
extern "C" void kernel_launch(void* const* d_in, const int* in_sizes, int n_in,
                              void* d_out, int out_size, void* d_ws, size_t ws_size,
                              hipStream_t stream) {
  (void)in_sizes; (void)n_in; (void)out_size; (void)ws_size;
  // JAX pytree flatten order (dict keys sorted at every level):
  // hist, lanes, params{dec{bhh,bih,whh,wih}, dyn_emb{b,w}, enc_hist, enc_l1,
  // enc_l2, enc_l3 (each {bhh,bih,whh,wih}), ip_emb{b,w}, lane_emb1..3{b,w}, out{b,w}}
  const float* hist  = (const float*)d_in[0];
  const float* lanes = (const float*)d_in[1];
  int i = 2;
  const float *dec_bhh[4], *dec_bih[4], *dec_whh[4], *dec_wih[4];
  for (int l = 0; l < 4; ++l) dec_bhh[l] = (const float*)d_in[i++];
  for (int l = 0; l < 4; ++l) dec_bih[l] = (const float*)d_in[i++];
  for (int l = 0; l < 4; ++l) dec_whh[l] = (const float*)d_in[i++];
  for (int l = 0; l < 4; ++l) dec_wih[l] = (const float*)d_in[i++];
  const float* dyn_b = (const float*)d_in[i++];
  const float* dyn_w = (const float*)d_in[i++];
  const float *e_bhh[4][4], *e_bih[4][4], *e_whh[4][4], *e_wih[4][4];
  for (int e = 0; e < 4; ++e) {
    for (int l = 0; l < 4; ++l) e_bhh[e][l] = (const float*)d_in[i++];
    for (int l = 0; l < 4; ++l) e_bih[e][l] = (const float*)d_in[i++];
    for (int l = 0; l < 4; ++l) e_whh[e][l] = (const float*)d_in[i++];
    for (int l = 0; l < 4; ++l) e_wih[e][l] = (const float*)d_in[i++];
  }
  const float* ip_b = (const float*)d_in[i++];
  const float* ip_w = (const float*)d_in[i++];
  const float *le_b[3], *le_w[3];
  for (int j = 0; j < 3; ++j) {
    le_b[j] = (const float*)d_in[i++];
    le_w[j] = (const float*)d_in[i++];
  }
  const float* out_b = (const float*)d_in[i++];
  const float* out_w = (const float*)d_in[i++];

  _Float16* enc_buf = (_Float16*)d_ws;   // [8192][64] f16 = 1 MB

  EncArgs ea;
  ea.x[0] = hist; ea.T[0] = 30; ea.emb_w[0] = ip_w; ea.emb_b[0] = ip_b;
  for (int j = 0; j < 3; ++j) {
    ea.x[1 + j] = lanes + (size_t)j * 100 * BTOT * 2;
    ea.T[1 + j] = 100;
    ea.emb_w[1 + j] = le_w[j];
    ea.emb_b[1 + j] = le_b[j];
  }
  for (int e = 0; e < 4; ++e)
    for (int l = 0; l < 4; ++l) {
      ea.wih[e][l] = e_wih[e][l]; ea.whh[e][l] = e_whh[e][l];
      ea.bih[e][l] = e_bih[e][l]; ea.bhh[e][l] = e_bhh[e][l];
    }
  ea.dyn_w = dyn_w; ea.dyn_b = dyn_b; ea.enc_out = enc_buf;

  dim3 egrid(BTOT / 16, 4);
  size_t enc_lds = (size_t)(TMAX * 256 + 256) * sizeof(_Float16);  // 51712 B
  encoder_kernel<<<egrid, 32, enc_lds, stream>>>(ea);

  DecArgs da;
  da.enc = enc_buf;
  for (int l = 0; l < 4; ++l) {
    da.wih[l] = dec_wih[l]; da.whh[l] = dec_whh[l];
    da.bih[l] = dec_bih[l]; da.bhh[l] = dec_bhh[l];
  }
  da.out_w = out_w; da.out_b = out_b; da.out = (float*)d_out;

  size_t dec_lds = (size_t)(NFRAG * 512 + 4 * 512) * sizeof(_Float16); // 60416 B
  decoder_kernel<<<dim3(BTOT / 64), 128, dec_lds, stream>>>(da);
}
